// FrequencyEnhancement_84258668413311
// MI455X (gfx1250) — matrix operand order repaired and checked by reading, not compiled, not run on hardware
//
#include <hip/hip_runtime.h>

typedef __attribute__((ext_vector_type(16))) _Float16 v16h;
typedef __attribute__((ext_vector_type(8)))  _Float16 v8h;
typedef __attribute__((ext_vector_type(8)))  float    v8f;

#define ROWS  16384   // 32*512
#define N4K   4096
#define M1K   1024
#define ALPHA 1.0f

#define KSTEP   32                    // K per WMMA step
#define PSTEPS  8                     // K-steps per panel
#define PKSTEP  (KSTEP * PSTEPS)      // 256 K per panel
#define PELEMS  (64 * PKSTEP)         // 16384 f16 per panel
#define PBYTES  (PELEMS * 2)          // 32 KB per panel
#define SBYTES  4096                  // bytes per K-step slice in LDS

#define WMMA_F16(A, B, C) \
    __builtin_amdgcn_wmma_f32_16x16x32_f16(false, (A), false, (B), (short)0, (C), false, false)

// ---------------------------------------------------------------------------
// Kernel 1: low-frequency orthonormal DCT basis in f16 (exact integer phase).
// ---------------------------------------------------------------------------
__global__ void __launch_bounds__(256)
dct_basis_kernel(_Float16* __restrict__ Dmat, _Float16* __restrict__ Dt) {
    unsigned idx = blockIdx.x * 256u + threadIdx.x;   // 0 .. 1024*4096-1
    unsigned k = idx >> 12;
    unsigned n = idx & 4095u;
    unsigned p = ((2u * n + 1u) * k) & 16383u;        // phase mod 4N (pow2)
    float ang = (float)p * 3.83495196971e-4f;         // pi / 8192
    float s   = (k == 0u) ? 0.015625f : 0.02209708691207961f;
    _Float16 v = (_Float16)(s * cosf(ang));
    Dmat[(size_t)k * N4K + n] = v;
    Dt[(size_t)n * M1K + k]   = v;
}

// ---------------------------------------------------------------------------
// Kernel 1b: one streaming pass converting X (f32) -> Xh (f16).
// ---------------------------------------------------------------------------
__global__ void __launch_bounds__(256)
xh_convert_kernel(const float* __restrict__ X, _Float16* __restrict__ Xh) {
    size_t i = ((size_t)blockIdx.x * 256 + threadIdx.x) * 8;
    float4 f0 = *(const float4*)(X + i);
    float4 f1 = *(const float4*)(X + i + 4);
    v8h h;
    h[0] = (_Float16)f0.x; h[1] = (_Float16)f0.y; h[2] = (_Float16)f0.z; h[3] = (_Float16)f0.w;
    h[4] = (_Float16)f1.x; h[5] = (_Float16)f1.y; h[6] = (_Float16)f1.z; h[7] = (_Float16)f1.w;
    *(v8h*)(Xh + i) = h;
}

// ---------------------------------------------------------------------------
// Async panel staging: thread t issues PSTEPS GLOBAL_LOAD_ASYNC_TO_LDS_B128
// ops (one 16 B piece per K-step slice). LDS layout is fragment-native:
//   slice s, elem(((j*16+n)*2 + h)*16 + i) = D[nbase+j*16+n][kc + s*32 + (i/8)*16 + h*8 + i%8]
// so per K-step slice the per-thread LDS destination is linear t*16 bytes.
// ---------------------------------------------------------------------------
__device__ __forceinline__ void wait_async0() {
    asm volatile("s_wait_asynccnt 0" ::: "memory");
}

__device__ __forceinline__ void stage_panel(unsigned dstbase,
                                            const _Float16* __restrict__ src,
                                            int ldb, int nbase, int kc, int t) {
    const int jn = t >> 2;              // (j*16+n) in 0..63
    const int h  = (t >> 1) & 1;        // K half within a step
    const int i0 = (t & 1) * 8;
    const _Float16* g = src + (size_t)(nbase + jn) * ldb + kc + (h << 3) + (i0 << 1);
    const unsigned d = dstbase + (unsigned)(t * 16);
#pragma unroll
    for (int s = 0; s < PSTEPS; ++s) {
        asm volatile("global_load_async_to_lds_b128 %0, %1, off"
                     :: "v"(d + (unsigned)(s * SBYTES)), "v"(g + s * KSTEP)
                     : "memory");
    }
}

// A-fragment loaders (16-bit A 16x32 layout: row m, chunks half*8 and 16+half*8)
__device__ __forceinline__ v16h load_afrag(const _Float16* p) {
    v8h lo = *(const v8h*)p;
    v8h hi = *(const v8h*)(p + 16);
    return __builtin_shufflevector(lo, hi, 0,1,2,3,4,5,6,7,8,9,10,11,12,13,14,15);
}
__device__ __forceinline__ v16h load_afrag(const float* p) {
    float4 f0 = *(const float4*)(p);
    float4 f1 = *(const float4*)(p + 4);
    float4 f2 = *(const float4*)(p + 16);
    float4 f3 = *(const float4*)(p + 20);
    v16h a;
    a[0]  = (_Float16)f0.x; a[1]  = (_Float16)f0.y; a[2]  = (_Float16)f0.z; a[3]  = (_Float16)f0.w;
    a[4]  = (_Float16)f1.x; a[5]  = (_Float16)f1.y; a[6]  = (_Float16)f1.z; a[7]  = (_Float16)f1.w;
    a[8]  = (_Float16)f2.x; a[9]  = (_Float16)f2.y; a[10] = (_Float16)f2.z; a[11] = (_Float16)f2.w;
    a[12] = (_Float16)f3.x; a[13] = (_Float16)f3.y; a[14] = (_Float16)f3.z; a[15] = (_Float16)f3.w;
    return a;
}

// ---------------------------------------------------------------------------
// Kernel 2: Y = A @ Dmat^T -> f16.  A is f16 (Xh) or f32 (fallback).
// Block: 8 waves stacked in M (tile 256M x 64N); wave tile 32M x 64N.
// blockIdx.x sweeps N so the 256-row A strip stays hot in L2.
// Panels are double-buffered in LDS; inner 8-step loop is barrier-free.
// ---------------------------------------------------------------------------
template <typename AT>
__global__ void __launch_bounds__(256)
gemm1_kernel(const AT* __restrict__ Ain,
             const _Float16* __restrict__ Dmat,
             _Float16* __restrict__ Y) {
    __shared__ __align__(32) _Float16 bsh[2][PELEMS];   // 64 KB total

    const int t    = threadIdx.x;
    const int lane = t & 31;
    const int wave = t >> 5;
    const int half = lane >> 4;
    const int mr   = lane & 15;
    const int mbase = blockIdx.y * 256 + wave * 32;
    const int nbase = blockIdx.x * 64;
    const unsigned lds0 = (unsigned)(size_t)(&bsh[0][0]);

    v8f a00 = {}, a01 = {}, a02 = {}, a03 = {};
    v8f a10 = {}, a11 = {}, a12 = {}, a13 = {};

    const AT* arow0 = Ain + (size_t)(mbase + mr) * N4K;
    const AT* arow1 = Ain + (size_t)(mbase + 16 + mr) * N4K;
    const int boff = ((mr * 2) + half) * 16;

    stage_panel(lds0, Dmat, N4K, nbase, 0, t);
    wait_async0();
    __syncthreads();

    const int NP = N4K / PKSTEP;                 // 16 panels
    for (int pc = 0; pc < NP; ++pc) {
        const int kc = pc * PKSTEP;
        if (pc + 1 < NP)
            stage_panel(lds0 + (unsigned)(((pc + 1) & 1) * PBYTES),
                        Dmat, N4K, nbase, kc + PKSTEP, t);

        const _Float16* bb = bsh[pc & 1];
#pragma unroll 4
        for (int s = 0; s < PSTEPS; ++s) {
            const int k0 = kc + s * KSTEP;
            const _Float16* bs = bb + s * (SBYTES / 2);
            v16h w0 = *(const v16h*)(bs + 0 * 512 + boff);
            v16h w1 = *(const v16h*)(bs + 1 * 512 + boff);
            v16h w2 = *(const v16h*)(bs + 2 * 512 + boff);
            v16h w3 = *(const v16h*)(bs + 3 * 512 + boff);
            v16h a0 = load_afrag(arow0 + k0 + half * 8);
            v16h a1 = load_afrag(arow1 + k0 + half * 8);

            a00 = WMMA_F16(a0, w0, a00); a01 = WMMA_F16(a0, w1, a01);
            a02 = WMMA_F16(a0, w2, a02); a03 = WMMA_F16(a0, w3, a03);
            a10 = WMMA_F16(a1, w0, a10); a11 = WMMA_F16(a1, w1, a11);
            a12 = WMMA_F16(a1, w2, a12); a13 = WMMA_F16(a1, w3, a13);
        }

        if (pc + 1 < NP) wait_async0();
        __syncthreads();
    }

    _Float16* y0 = Y + (size_t)(mbase + half * 8) * M1K + (nbase + mr);
    _Float16* y1 = Y + (size_t)(mbase + 16 + half * 8) * M1K + (nbase + mr);
#pragma unroll
    for (int r = 0; r < 8; ++r) {
        y0[(size_t)r * M1K +  0] = (_Float16)a00[r];
        y0[(size_t)r * M1K + 16] = (_Float16)a01[r];
        y0[(size_t)r * M1K + 32] = (_Float16)a02[r];
        y0[(size_t)r * M1K + 48] = (_Float16)a03[r];
        y1[(size_t)r * M1K +  0] = (_Float16)a10[r];
        y1[(size_t)r * M1K + 16] = (_Float16)a11[r];
        y1[(size_t)r * M1K + 32] = (_Float16)a12[r];
        y1[(size_t)r * M1K + 48] = (_Float16)a13[r];
    }
}

// ---------------------------------------------------------------------------
// Kernel 3: X_low = Y @ D_low, fused epilogue out = x + g*(x - x_low).
// ---------------------------------------------------------------------------
__global__ void __launch_bounds__(256)
gemm2_idct(const _Float16* __restrict__ Yin,
           const _Float16* __restrict__ Dt,
           const float* __restrict__ X,
           const float* __restrict__ gate,
           float* __restrict__ Out) {
    __shared__ __align__(32) _Float16 bsh[2][PELEMS];   // 64 KB total

    const int t    = threadIdx.x;
    const int lane = t & 31;
    const int wave = t >> 5;
    const int half = lane >> 4;
    const int mr   = lane & 15;
    const int mbase = blockIdx.y * 256 + wave * 32;
    const int nbase = blockIdx.x * 64;
    const unsigned lds0 = (unsigned)(size_t)(&bsh[0][0]);

    v8f a00 = {}, a01 = {}, a02 = {}, a03 = {};
    v8f a10 = {}, a11 = {}, a12 = {}, a13 = {};

    const _Float16* arow0 = Yin + (size_t)(mbase + mr) * M1K;
    const _Float16* arow1 = Yin + (size_t)(mbase + 16 + mr) * M1K;
    const int boff = ((mr * 2) + half) * 16;

    stage_panel(lds0, Dt, M1K, nbase, 0, t);
    wait_async0();
    __syncthreads();

    const int NP = M1K / PKSTEP;                 // 4 panels
    for (int pc = 0; pc < NP; ++pc) {
        const int kc = pc * PKSTEP;
        if (pc + 1 < NP)
            stage_panel(lds0 + (unsigned)(((pc + 1) & 1) * PBYTES),
                        Dt, M1K, nbase, kc + PKSTEP, t);

        const _Float16* bb = bsh[pc & 1];
#pragma unroll 4
        for (int s = 0; s < PSTEPS; ++s) {
            const int k0 = kc + s * KSTEP;
            const _Float16* bs = bb + s * (SBYTES / 2);
            v16h w0 = *(const v16h*)(bs + 0 * 512 + boff);
            v16h w1 = *(const v16h*)(bs + 1 * 512 + boff);
            v16h w2 = *(const v16h*)(bs + 2 * 512 + boff);
            v16h w3 = *(const v16h*)(bs + 3 * 512 + boff);
            v16h a0 = load_afrag(arow0 + k0 + half * 8);
            v16h a1 = load_afrag(arow1 + k0 + half * 8);

            a00 = WMMA_F16(a0, w0, a00); a01 = WMMA_F16(a0, w1, a01);
            a02 = WMMA_F16(a0, w2, a02); a03 = WMMA_F16(a0, w3, a03);
            a10 = WMMA_F16(a1, w0, a10); a11 = WMMA_F16(a1, w1, a11);
            a12 = WMMA_F16(a1, w2, a12); a13 = WMMA_F16(a1, w3, a13);
        }

        if (pc + 1 < NP) wait_async0();
        __syncthreads();
    }

    const int col = nbase + mr;
#pragma unroll
    for (int r = 0; r < 8; ++r) {
        const int row = mbase + half * 8 + r;
        const float g = ALPHA * gate[row];
        const size_t base = (size_t)row * N4K + col;
        float xv;
        xv = X[base +  0]; Out[base +  0] = xv + g * (xv - a00[r]);
        xv = X[base + 16]; Out[base + 16] = xv + g * (xv - a01[r]);
        xv = X[base + 32]; Out[base + 32] = xv + g * (xv - a02[r]);
        xv = X[base + 48]; Out[base + 48] = xv + g * (xv - a03[r]);
    }
#pragma unroll
    for (int r = 0; r < 8; ++r) {
        const int row = mbase + 16 + half * 8 + r;
        const float g = ALPHA * gate[row];
        const size_t base = (size_t)row * N4K + col;
        float xv;
        xv = X[base +  0]; Out[base +  0] = xv + g * (xv - a10[r]);
        xv = X[base + 16]; Out[base + 16] = xv + g * (xv - a11[r]);
        xv = X[base + 32]; Out[base + 32] = xv + g * (xv - a12[r]);
        xv = X[base + 48]; Out[base + 48] = xv + g * (xv - a13[r]);
    }
}

// ---------------------------------------------------------------------------
extern "C" void kernel_launch(void* const* d_in, const int* in_sizes, int n_in,
                              void* d_out, int out_size, void* d_ws, size_t ws_size,
                              hipStream_t stream) {
    (void)in_sizes; (void)n_in; (void)out_size;

    const float* x    = (const float*)d_in[0];   // [32,512,4096] f32
    const float* gate = (const float*)d_in[1];   // [32,512] f32
    float* out        = (float*)d_out;           // [32,512,4096] f32

    // Workspace layout: Dmat 8MB | Dt 8MB | Y 32MB | (optional) Xh 128MB
    const size_t offDt = (size_t)M1K * N4K * 2;
    const size_t offY  = 2 * offDt;
    const size_t offXh = offY + (size_t)ROWS * M1K * 2;
    const size_t need  = offXh + (size_t)ROWS * N4K * 2;

    _Float16* Dmat = (_Float16*)d_ws;
    _Float16* Dt   = (_Float16*)((char*)d_ws + offDt);
    _Float16* Yw   = (_Float16*)((char*)d_ws + offY);
    _Float16* Xh   = (_Float16*)((char*)d_ws + offXh);

    dct_basis_kernel<<<(M1K * N4K) / 256, 256, 0, stream>>>(Dmat, Dt);

    dim3 g1(M1K / 64, ROWS / 256);   // grid.x = N tiles (fast) -> L2 reuse of A strip
    if (ws_size >= need) {
        xh_convert_kernel<<<((size_t)ROWS * N4K) / (256 * 8), 256, 0, stream>>>(x, Xh);
        gemm1_kernel<_Float16><<<g1, 256, 0, stream>>>(Xh, Dmat, Yw);
    } else {
        gemm1_kernel<float><<<g1, 256, 0, stream>>>(x, Dmat, Yw);
    }

    dim3 g2(N4K / 64, ROWS / 256);
    gemm2_idct<<<g2, 256, 0, stream>>>(Yw, Dt, x, gate, out);
}
